// TransformerDecoderBlock_11519102287885
// MI455X (gfx1250) — matrix operand order repaired and checked by reading, not compiled, not run on hardware
//
#include <hip/hip_runtime.h>
#include <hip/hip_bf16.h>

typedef __attribute__((ext_vector_type(16))) _Float16 v16h;
typedef __attribute__((ext_vector_type(8)))  _Float16 v8h;
typedef __attribute__((ext_vector_type(8)))  float    v8f;

union Frag { v16h v; v8h h[2]; };

// Async global->LDS 16-byte copy (CDNA5 GLOBAL_LOAD_ASYNC_TO_LDS_B128).
#define ASYNC_B128(ldsOff, gptr)                                            \
  asm volatile("global_load_async_to_lds_b128 %0, %1, off"                  \
               :: "v"(ldsOff), "v"(gptr) : "memory")
#define WAIT_ASYNC_8() asm volatile("s_wait_asynccnt 0x8" ::: "memory")
#define WAIT_ASYNC_0() asm volatile("s_wait_asynccnt 0x0" ::: "memory")

// ---------------------------------------------------------------------------
// Tiled transpose + fp32->fp16 cast:  wT[n][k] = (f16) w[k][n]
// ---------------------------------------------------------------------------
__global__ __launch_bounds__(256) void transpose_cast(
    const float* __restrict__ w, _Float16* __restrict__ wt, int K, int N) {
  __shared__ float tile[32][33];
  const int k0 = blockIdx.x * 32;
  const int n0 = blockIdx.y * 32;
  const int tx = threadIdx.x;      // 0..31
  const int ty = threadIdx.y;      // 0..7
#pragma unroll
  for (int j = 0; j < 4; ++j)
    tile[ty + 8 * j][tx] = w[(size_t)(k0 + ty + 8 * j) * N + n0 + tx];
  __syncthreads();
#pragma unroll
  for (int j = 0; j < 4; ++j)
    wt[(size_t)(n0 + ty + 8 * j) * K + k0 + tx] = (_Float16)tile[tx][ty + 8 * j];
}

// ---------------------------------------------------------------------------
// LayerNorm over C=1024 + cast to f16. One block (256 thr) per row.
// ---------------------------------------------------------------------------
__global__ __launch_bounds__(256) void ln_cast(
    const float* __restrict__ x, const float* __restrict__ g,
    const float* __restrict__ bb, _Float16* __restrict__ out) {
  __shared__ float r1[256], r2[256];
  const int t = threadIdx.x;
  const size_t base = (size_t)blockIdx.x * 1024;
  float v[4];
  float s = 0.f, sq = 0.f;
#pragma unroll
  for (int i = 0; i < 4; ++i) {
    v[i] = x[base + t + 256 * i];
    s += v[i];
    sq += v[i] * v[i];
  }
  r1[t] = s; r2[t] = sq;
  __syncthreads();
  for (int off = 128; off > 0; off >>= 1) {
    if (t < off) { r1[t] += r1[t + off]; r2[t] += r2[t + off]; }
    __syncthreads();
  }
  const float mu   = r1[0] * (1.0f / 1024.0f);
  const float var  = r2[0] * (1.0f / 1024.0f) - mu * mu;
  const float rstd = rsqrtf(var + 1e-5f);
#pragma unroll
  for (int i = 0; i < 4; ++i) {
    const int idx = t + 256 * i;
    out[base + idx] = (_Float16)((v[i] - mu) * rstd * g[idx] + bb[idx]);
  }
}

// ---------------------------------------------------------------------------
// 128x128 WMMA GEMM, BK=64, double-buffered async global->LDS staging.
// A: f16 [M][K] row-major. Bt: f16 [N][K] row-major.
// 256 threads = 8 waves; wave (m,n) grid = 2x4; wave tile = 64x32.
// Per stage: 16 WMMAs/wave behind one async DMA stage (8 b128/thread).
// ---------------------------------------------------------------------------
template <bool GELU, bool RES, bool OUT32, bool OUT16>
__global__ __launch_bounds__(256) void gemm128(
    const _Float16* __restrict__ A, const _Float16* __restrict__ Bt,
    const float* __restrict__ bias, const float* __restrict__ res,
    float* __restrict__ out32, _Float16* __restrict__ out16,
    int M, int N, int K) {
  constexpr int LD = 72;                         // 64 + 8 pad (16B aligned rows)
  __shared__ _Float16 sAB[2][2][128 * LD];       // [buf][A/B][...]  (72 KB)
  const int t    = threadIdx.x;
  const int lane = t & 31;
  const int wave = t >> 5;
  const int wm   = (wave & 1) * 64;
  const int wn   = (wave >> 1) * 32;
  const int hi   = lane >> 4;
  const int lr   = lane & 15;
  const int m0   = blockIdx.y * 128;
  const int n0   = blockIdx.x * 128;

  // per-thread staging slots: 1024 16B chunks per 128x64 tile, 4 per thread
  const _Float16* gA[4];
  const _Float16* gB[4];
  unsigned lA[4], lB[4];
#pragma unroll
  for (int i = 0; i < 4; ++i) {
    const int c = t + i * 256;
    const int r = c >> 3, q = (c & 7) * 8;
    gA[i] = A + (size_t)(m0 + r) * K + q;
    gB[i] = Bt + (size_t)(n0 + r) * K + q;
    lA[i] = (unsigned)(size_t)&sAB[0][0][r * LD + q];
    lB[i] = (unsigned)(size_t)&sAB[0][1][r * LD + q];
  }
  const unsigned bufStride =
      (unsigned)(2u * 128u * LD * (unsigned)sizeof(_Float16));

  auto issueStage = [&](int buf, int k0) {
    const unsigned d = (unsigned)buf * bufStride;
#pragma unroll
    for (int i = 0; i < 4; ++i) {
      ASYNC_B128(lA[i] + d, gA[i] + k0);
      ASYNC_B128(lB[i] + d, gB[i] + k0);
    }
  };

  v8f acc[4][2] = {};
  const int nsteps = K / 64;
  issueStage(0, 0);

  for (int s = 0; s < nsteps; ++s) {
    const int buf = s & 1;
    if (s + 1 < nsteps) {
      issueStage(buf ^ 1, (s + 1) * 64);
      WAIT_ASYNC_8();                 // in-order: the 8 oldest (stage s) done
    } else {
      WAIT_ASYNC_0();
    }
    __syncthreads();                  // stage-s data visible to all waves

    const _Float16* pA = sAB[buf][0];
    const _Float16* pB = sAB[buf][1];
#pragma unroll
    for (int kk = 0; kk < 2; ++kk) {
      Frag aF[4], bF[2];
#pragma unroll
      for (int mt = 0; mt < 4; ++mt) {
        const int base = (wm + mt * 16 + lr) * LD + kk * 32 + hi * 8;
        aF[mt].h[0] = *(const v8h*)&pA[base];
        aF[mt].h[1] = *(const v8h*)&pA[base + 16];
      }
#pragma unroll
      for (int nt = 0; nt < 2; ++nt) {
        const int base = (wn + nt * 16 + lr) * LD + kk * 32 + hi * 8;
        bF[nt].h[0] = *(const v8h*)&pB[base];
        bF[nt].h[1] = *(const v8h*)&pB[base + 16];
      }
#pragma unroll
      for (int mt = 0; mt < 4; ++mt)
#pragma unroll
        for (int nt = 0; nt < 2; ++nt)
          acc[mt][nt] = __builtin_amdgcn_wmma_f32_16x16x32_f16(
              false, aF[mt].v, false, bF[nt].v, (short)0, acc[mt][nt],
              false, false);
    }
    __syncthreads();                  // all reads of buf done before re-issue
  }

#pragma unroll
  for (int mt = 0; mt < 4; ++mt)
#pragma unroll
    for (int nt = 0; nt < 2; ++nt)
#pragma unroll
      for (int i = 0; i < 8; ++i) {
        const int gr = m0 + wm + mt * 16 + i + hi * 8;
        const int gc = n0 + wn + nt * 16 + lr;
        float v = acc[mt][nt][i] + bias[gc];
        if constexpr (GELU) v = 0.5f * v * (1.0f + erff(v * 0.70710678118f));
        if constexpr (RES)  v += res[(size_t)gr * N + gc];
        if constexpr (OUT32) out32[(size_t)gr * N + gc] = v;
        if constexpr (OUT16) out16[(size_t)gr * N + gc] = (_Float16)v;
      }
}

// ---------------------------------------------------------------------------
// Causal flash attention. Block = 128 thr (4 waves) handles one
// (b, h, 64-query tile). Online softmax in f32, WMMA for QK^T and PV.
// Q/K tiles staged with async global->LDS; V staged transposed manually.
// K/V global pointers are hoisted and incremented per key-tile.
// qkv: f16 [B*T][3C], aout: f16 [B*T][C]. scale = C^-0.5 = 1/32.
// ---------------------------------------------------------------------------
__global__ __launch_bounds__(128) void attn_kernel(
    const _Float16* __restrict__ qkv, _Float16* __restrict__ aout) {
  constexpr int LD = 72;                       // 64 + 8 pad
  __shared__ _Float16 sQ[64 * LD];
  __shared__ _Float16 sK[64 * LD];
  __shared__ _Float16 sVt[64 * LD];            // [d][key]
  __shared__ _Float16 sP[64 * LD];
  __shared__ float    sS[64 * LD];
  __shared__ float    sM[64], sL[64], sAlpha[64];

  const int t = threadIdx.x;
  const int lane = t & 31, w = t >> 5, hi = lane >> 4, lr = lane & 15;
  const int qt = blockIdx.x, h = blockIdx.y, b = blockIdx.z;
  const int qBase = qt * 64;
  const size_t rs = 3 * 1024;                  // qkv row stride (3C)
  const size_t kvStep = 64 * rs;               // advance 64 keys per tile
  const _Float16* qkvB = qkv + (size_t)b * 2048 * rs + h * 64;

  // per-thread staging slot for 64x64 tiles: 512 chunks, 4 per thread
  int srow[4], scol[4];
  unsigned lK[4];
  const _Float16* gK[4];
  const _Float16* gV[4];
#pragma unroll
  for (int i = 0; i < 4; ++i) {
    const int c = t + i * 128;
    srow[i] = c >> 3;
    scol[i] = (c & 7) * 8;
    lK[i] = (unsigned)(size_t)&sK[srow[i] * LD + scol[i]];
    gK[i] = qkvB + (size_t)srow[i] * rs + 1024 + scol[i];
    gV[i] = qkvB + (size_t)srow[i] * rs + 2048 + scol[i];
  }

  // stage Q (async)
#pragma unroll
  for (int i = 0; i < 4; ++i) {
    const unsigned lq = (unsigned)(size_t)&sQ[srow[i] * LD + scol[i]];
    ASYNC_B128(lq, &qkvB[(size_t)(qBase + srow[i]) * rs + scol[i]]);
  }
  if (t < 64) { sM[t] = -1e30f; sL[t] = 0.f; }
  WAIT_ASYNC_0();
  __syncthreads();

  Frag aQ[2];
#pragma unroll
  for (int kk = 0; kk < 2; ++kk) {
    const int base = (w * 16 + lr) * LD + kk * 32 + hi * 8;
    aQ[kk].h[0] = *(const v8h*)&sQ[base];
    aQ[kk].h[1] = *(const v8h*)&sQ[base + 16];
  }

  v8f accO[4] = {};

  for (int kt = 0; kt <= qt; ++kt) {
    const int kBase = kt * 64;
    // stage K (async, row-major [key][d] == B-layout for QK^T)
#pragma unroll
    for (int i = 0; i < 4; ++i) ASYNC_B128(lK[i], gK[i]);
    // stage V transposed: sVt[d][key]
#pragma unroll
    for (int i = 0; i < 4; ++i) {
      const v8h vv = *(const v8h*)gV[i];
#pragma unroll
      for (int j = 0; j < 8; ++j) sVt[(scol[i] + j) * LD + srow[i]] = vv[j];
    }
#pragma unroll
    for (int i = 0; i < 4; ++i) { gK[i] += kvStep; gV[i] += kvStep; }
    WAIT_ASYNC_0();
    __syncthreads();

    // S = (Q K^T) * scale, causal mask, write f32 to LDS
#pragma unroll
    for (int nt = 0; nt < 4; ++nt) {
      v8f s = {};
#pragma unroll
      for (int kk = 0; kk < 2; ++kk) {
        Frag bK;
        const int base = (nt * 16 + lr) * LD + kk * 32 + hi * 8;
        bK.h[0] = *(const v8h*)&sK[base];
        bK.h[1] = *(const v8h*)&sK[base + 16];
        s = __builtin_amdgcn_wmma_f32_16x16x32_f16(
            false, aQ[kk].v, false, bK.v, (short)0, s, false, false);
      }
#pragma unroll
      for (int i = 0; i < 8; ++i) {
        const int qi = qBase + w * 16 + i + hi * 8;
        const int ki = kBase + nt * 16 + lr;
        float val = s[i] * 0.03125f;           // C^-0.5
        if (ki > qi) val = -1e30f;
        sS[(w * 16 + i + hi * 8) * LD + nt * 16 + lr] = val;
      }
    }
    __syncthreads();

    // per-row online softmax (one thread per row)
    if (t < 64) {
      const float mold = sM[t];
      float mx = mold;
      for (int j = 0; j < 64; ++j) mx = fmaxf(mx, sS[t * LD + j]);
      const float alpha = __expf(mold - mx);
      float sum = 0.f;
      for (int j = 0; j < 64; ++j) {
        const float p = __expf(sS[t * LD + j] - mx);
        sum += p;
        sP[t * LD + j] = (_Float16)p;
      }
      sM[t] = mx;
      sL[t] = sL[t] * alpha + sum;
      sAlpha[t] = alpha;
    }
    __syncthreads();

    // O = alpha * O + P @ V
    float al[8];
#pragma unroll
    for (int i = 0; i < 8; ++i) al[i] = sAlpha[w * 16 + i + hi * 8];
#pragma unroll
    for (int nt = 0; nt < 4; ++nt)
#pragma unroll
      for (int i = 0; i < 8; ++i) accO[nt][i] *= al[i];

#pragma unroll
    for (int kk = 0; kk < 2; ++kk) {
      Frag aP;
      const int base = (w * 16 + lr) * LD + kk * 32 + hi * 8;
      aP.h[0] = *(const v8h*)&sP[base];
      aP.h[1] = *(const v8h*)&sP[base + 16];
#pragma unroll
      for (int nt = 0; nt < 4; ++nt) {
        Frag bV;
        const int vb = (nt * 16 + lr) * LD + kk * 32 + hi * 8;
        bV.h[0] = *(const v8h*)&sVt[vb];
        bV.h[1] = *(const v8h*)&sVt[vb + 16];
        accO[nt] = __builtin_amdgcn_wmma_f32_16x16x32_f16(
            false, aP.v, false, bV.v, (short)0, accO[nt], false, false);
      }
    }
    __syncthreads();
  }

  // normalize and write [B*T][C] f16
  float inv[8];
#pragma unroll
  for (int i = 0; i < 8; ++i) inv[i] = 1.0f / sL[w * 16 + i + hi * 8];
  _Float16* outB = aout + (size_t)b * 2048 * 1024 + h * 64;
#pragma unroll
  for (int nt = 0; nt < 4; ++nt)
#pragma unroll
    for (int i = 0; i < 8; ++i) {
      const int row = qBase + w * 16 + i + hi * 8;
      outB[(size_t)row * 1024 + nt * 16 + lr] = (_Float16)(accO[nt][i] * inv[i]);
    }
}

// ---------------------------------------------------------------------------
extern "C" void kernel_launch(void* const* d_in, const int* in_sizes, int n_in,
                              void* d_out, int out_size, void* d_ws, size_t ws_size,
                              hipStream_t stream) {
  (void)in_sizes; (void)n_in; (void)out_size; (void)ws_size;
  const float* x           = (const float*)d_in[0];
  const float* ln1_g       = (const float*)d_in[1];
  const float* ln1_b       = (const float*)d_in[2];
  const float* w_qkv       = (const float*)d_in[3];
  const float* b_qkv       = (const float*)d_in[4];
  const float* w_attn_proj = (const float*)d_in[5];
  const float* b_attn_proj = (const float*)d_in[6];
  const float* ln2_g       = (const float*)d_in[7];
  const float* ln2_b       = (const float*)d_in[8];
  const float* w_fc        = (const float*)d_in[9];
  const float* b_fc        = (const float*)d_in[10];
  const float* w_out       = (const float*)d_in[11];
  const float* b_out       = (const float*)d_in[12];

  constexpr int M = 4096;  // B*T
  char* ws = (char*)d_ws;
  size_t off = 0;
  auto alloc = [&](size_t bytes) -> void* {
    void* p = ws + off;
    off += (bytes + 255) & ~(size_t)255;
    return p;
  };
  _Float16* wqkvT  = (_Float16*)alloc((size_t)3072 * 1024 * 2);
  _Float16* wprojT = (_Float16*)alloc((size_t)1024 * 1024 * 2);
  _Float16* wfcT   = (_Float16*)alloc((size_t)4096 * 1024 * 2);
  _Float16* woutT  = (_Float16*)alloc((size_t)1024 * 4096 * 2);
  _Float16* h1     = (_Float16*)alloc((size_t)M * 1024 * 2);
  _Float16* qkvbuf = (_Float16*)alloc((size_t)M * 3072 * 2);
  _Float16* attnO  = (_Float16*)alloc((size_t)M * 1024 * 2);
  float*    x1     = (float*)   alloc((size_t)M * 1024 * 4);
  _Float16* h2     = (_Float16*)alloc((size_t)M * 1024 * 2);
  _Float16* fcbuf  = (_Float16*)alloc((size_t)M * 4096 * 2);

  const dim3 tb(32, 8);
  transpose_cast<<<dim3(1024 / 32, 3072 / 32), tb, 0, stream>>>(w_qkv, wqkvT, 1024, 3072);
  transpose_cast<<<dim3(1024 / 32, 1024 / 32), tb, 0, stream>>>(w_attn_proj, wprojT, 1024, 1024);
  transpose_cast<<<dim3(1024 / 32, 4096 / 32), tb, 0, stream>>>(w_fc, wfcT, 1024, 4096);
  transpose_cast<<<dim3(4096 / 32, 1024 / 32), tb, 0, stream>>>(w_out, woutT, 4096, 1024);

  // h1 = f16(LN1(x))
  ln_cast<<<M, 256, 0, stream>>>(x, ln1_g, ln1_b, h1);
  // qkv = h1 @ w_qkv + b_qkv   (f16 out)
  gemm128<false, false, false, true><<<dim3(3072 / 128, M / 128), 256, 0, stream>>>(
      h1, wqkvT, b_qkv, nullptr, nullptr, qkvbuf, M, 3072, 1024);
  // causal MHA
  attn_kernel<<<dim3(2048 / 64, 16, 2), 128, 0, stream>>>(qkvbuf, attnO);
  // x1 = x + attnO @ w_attn_proj + b_attn_proj   (f32 out)
  gemm128<false, true, true, false><<<dim3(1024 / 128, M / 128), 256, 0, stream>>>(
      attnO, wprojT, b_attn_proj, x, x1, nullptr, M, 1024, 1024);
  // h2 = f16(LN2(x1))
  ln_cast<<<M, 256, 0, stream>>>(x1, ln2_g, ln2_b, h2);
  // fc = gelu(h2 @ w_fc + b_fc)   (f16 out)
  gemm128<true, false, false, true><<<dim3(4096 / 128, M / 128), 256, 0, stream>>>(
      h2, wfcT, b_fc, nullptr, nullptr, fcbuf, M, 4096, 1024);
  // out = x1 + fc @ w_out + b_out   (f32 out)
  gemm128<false, true, true, false><<<dim3(1024 / 128, M / 128), 256, 0, stream>>>(
      fcbuf, woutT, b_out, x1, (float*)d_out, nullptr, M, 1024, 4096);
}
